// GNN_GCNConv_homogen_basic_12378095747343
// MI455X (gfx1250) — matrix operand order repaired and checked by reading, not compiled, not run on hardware
//
#include <hip/hip_runtime.h>
#include <hip/hip_bf16.h>

// MI455X (gfx1250) implementation.
//  - Fused weight W1@Wc computed on device, packed directly into the CDNA5
//    WMMA B-fragment layout (bf16).
//  - Main GEMM x@Wf uses v_wmma_f32_16x16x32_bf16 (fp32 accumulate);
//    weights staged to LDS via inline-asm global_load_async_to_lds_b128
//    (ASYNCcnt / s_wait_asynccnt), zero-init accumulators (inline SRC2=0),
//    bias added at store.
//  - GCN aggregation via integer-degree + rsqrt norm, wave-per-edge
//    gather + global fp32 atomics (all L2-resident: feature mats are 25.6MB
//    each vs 192MB L2).
//  - Edge scoring: wave-per-edge dot product + shuffle reduction (wave32).

typedef __bf16 bf16_t;
typedef __attribute__((ext_vector_type(16))) __bf16 v16bf;
typedef __attribute__((ext_vector_type(8)))  float  v8f;

#define N_NODES 50000
#define D_IN    256
#define D_OUT   128
#define N_POS   1600000
#define N_PRED  500000

#define WF_ELEMS (8 * 8 * 32 * 16)   /* 32768 bf16 = 64 KB */

// ---------------------------------------------------------------------------
// Wf[k][n] = sum_j W1[k][j] * Wc[j][n]; store bf16 in WMMA B-fragment order.
// B 32x16 (KxN) bf16 fragment as packed here (contiguous k): lane l holds col N=l&15, kbase=(l>>4)*16,
// element e (0..15) = K kbase+e.  Fragment id = (n>>4)*8 + (k>>5).
// ---------------------------------------------------------------------------
__global__ void k_fuse_w(const float* __restrict__ W1, const float* __restrict__ Wc,
                         bf16_t* __restrict__ wfrag) {
    int idx = blockIdx.x * blockDim.x + threadIdx.x;   // 0..32767
    int k = idx >> 7;                                  // input dim 0..255
    int n = idx & 127;                                 // output col 0..127
    float acc = 0.f;
    for (int j = 0; j < D_IN; ++j)
        acc = fmaf(W1[k * D_IN + j], Wc[j * D_OUT + n], acc);
    int t = n >> 4, nn = n & 15;
    int s = k >> 5, k32 = k & 31;
    int lane = nn + ((k32 >> 4) << 4);
    int e = k32 & 15;
    wfrag[(((t * 8 + s) * 32 + lane) << 4) + e] = (bf16_t)acc;
}

__global__ void k_fuse_b(const float* __restrict__ b1, const float* __restrict__ Wc,
                         float* __restrict__ hb) {
    int n = threadIdx.x;                               // 0..127
    float acc = 0.f;
    for (int j = 0; j < D_IN; ++j)
        acc = fmaf(b1[j], Wc[j * D_OUT + n], acc);
    hb[n] = acc;
}

// ---------------------------------------------------------------------------
// h[m][n] = sum_k x[m][k] * Wf[k][n] + hb[n]
// One wave per 16-row tile; 8 N-tiles of 16; K=256 = 8 steps of 32.
// A 16x32 bf16 fragment as built here: lane l holds row M=l&15, kb=(l>>4)*16,
// elems 0..15 = K kb..kb+15 (same contiguous k order as the packed B).
// ---------------------------------------------------------------------------
__global__ void __launch_bounds__(256) k_gemm(const float* __restrict__ x,
                                              const bf16_t* __restrict__ wfrag,
                                              const float* __restrict__ hb,
                                              float* __restrict__ h) {
    __shared__ __align__(16) bf16_t lwf[WF_ELEMS];     // 64 KB per workgroup

    // Cooperative async stage of the packed weight fragments into LDS.
    // VDST = LDS byte address (low 32 bits of the flat LDS-aperture address,
    // per ISA 10.2); VADDR = 64-bit global address;
    // SADDR = off (GV mode). Tracked by ASYNCcnt.
    {
        const char* src = (const char*)wfrag;
        char* dst = (char*)lwf;
#pragma unroll
        for (int it = 0; it < 16; ++it) {
            int off = (it * 256 + (int)threadIdx.x) * 16;
            unsigned long long ga = (unsigned long long)(src + off);
            unsigned la = (unsigned)(unsigned long long)(dst + off);
            asm volatile("global_load_async_to_lds_b128 %0, %1, off"
                         :: "v"(la), "v"(ga)
                         : "memory");
        }
        asm volatile("s_wait_asynccnt 0x0" ::: "memory");
        __syncthreads();
    }

    int lane = threadIdx.x & 31;
    int wv   = threadIdx.x >> 5;
    int mt   = blockIdx.x * 8 + wv;
    if (mt * 16 < N_NODES) {
        int m0   = mt * 16;
        int lrow = lane & 15;
        int lhi  = lane >> 4;

        v8f acc[8];
#pragma unroll
        for (int t = 0; t < 8; ++t)
            acc[t] = (v8f){0.f, 0.f, 0.f, 0.f, 0.f, 0.f, 0.f, 0.f};

        const float* xr = x + (size_t)(m0 + lrow) * D_IN + lhi * 16;
#pragma unroll
        for (int s = 0; s < 8; ++s) {
            const float4* p0 = (const float4*)(xr + 32 * s);
            const float4* p1 = (const float4*)(xr + 32 * s + 8);
            float4 c0 = p0[0], c1 = p0[1];
            float4 c2 = p1[0], c3 = p1[1];
            v16bf a;
            a[0]  = (bf16_t)c0.x; a[1]  = (bf16_t)c0.y; a[2]  = (bf16_t)c0.z; a[3]  = (bf16_t)c0.w;
            a[4]  = (bf16_t)c1.x; a[5]  = (bf16_t)c1.y; a[6]  = (bf16_t)c1.z; a[7]  = (bf16_t)c1.w;
            a[8]  = (bf16_t)c2.x; a[9]  = (bf16_t)c2.y; a[10] = (bf16_t)c2.z; a[11] = (bf16_t)c2.w;
            a[12] = (bf16_t)c3.x; a[13] = (bf16_t)c3.y; a[14] = (bf16_t)c3.z; a[15] = (bf16_t)c3.w;
#pragma unroll
            for (int t = 0; t < 8; ++t) {
                v16bf b = *(const v16bf*)(lwf + (((t * 8 + s) * 32 + lane) << 4));
                acc[t] = __builtin_amdgcn_wmma_f32_16x16x32_bf16(
                    false, a, false, b, (short)0, acc[t], false, false);
            }
        }

        // C/D layout: lane l, VGPR j -> D[M = j + 8*(l>>4)][N = l&15]
#pragma unroll
        for (int t = 0; t < 8; ++t) {
            float bias = hb[t * 16 + lrow];
#pragma unroll
            for (int j = 0; j < 8; ++j) {
                h[(size_t)(m0 + j + 8 * lhi) * D_OUT + t * 16 + lrow] = acc[t][j] + bias;
            }
        }
    }
}

// ---------------------------------------------------------------------------
// Degree / normalization
// ---------------------------------------------------------------------------
__global__ void k_deg_init(unsigned* __restrict__ cnt) {
    int i = blockIdx.x * blockDim.x + threadIdx.x;
    if (i < N_NODES) cnt[i] = 1u;                      // self-loop
}

__global__ void k_deg_acc(const int* __restrict__ col, unsigned* __restrict__ cnt) {
    int e = blockIdx.x * blockDim.x + threadIdx.x;
    if (e < N_POS) atomicAdd(&cnt[col[e]], 1u);
}

__global__ void k_dinv(const unsigned* __restrict__ cnt, float* __restrict__ dinv) {
    int i = blockIdx.x * blockDim.x + threadIdx.x;
    if (i < N_NODES) dinv[i] = rsqrtf((float)cnt[i]);
}

// g = h * dinv[node]  (in place over h), and init acc = g (self-loop term).
__global__ void k_scale(float* __restrict__ h, float* __restrict__ acc,
                        const float* __restrict__ dinv) {
    int i4 = blockIdx.x * blockDim.x + threadIdx.x;    // float4 index
    if (i4 >= N_NODES * 32) return;
    int node = i4 >> 5;
    float d = dinv[node];
    float4 v = ((const float4*)h)[i4];
    v.x *= d; v.y *= d; v.z *= d; v.w *= d;
    ((float4*)h)[i4]   = v;
    ((float4*)acc)[i4] = v;
}

// One wave per edge: acc[col] += g[row]  (128 floats, 4 per lane).
__global__ void __launch_bounds__(256) k_scatter(const int* __restrict__ row,
                                                 const int* __restrict__ col,
                                                 const float* __restrict__ g,
                                                 float* __restrict__ acc) {
    int e = blockIdx.x * 8 + (threadIdx.x >> 5);
    if (e >= N_POS) return;
    int lane = threadIdx.x & 31;
    int r = row[e], c = col[e];
    float4 v = ((const float4*)g)[r * 32 + lane];
    float* dst = acc + (size_t)c * D_OUT + lane * 4;
    unsafeAtomicAdd(dst + 0, v.x);
    unsafeAtomicAdd(dst + 1, v.y);
    unsafeAtomicAdd(dst + 2, v.z);
    unsafeAtomicAdd(dst + 3, v.w);
}

// out = acc * dinv[node] + bc   (in place over acc)
__global__ void k_finish(float* __restrict__ acc, const float* __restrict__ dinv,
                         const float* __restrict__ bc) {
    int i4 = blockIdx.x * blockDim.x + threadIdx.x;
    if (i4 >= N_NODES * 32) return;
    int node = i4 >> 5, c4 = i4 & 31;
    float d = dinv[node];
    float4 v = ((const float4*)acc)[i4];
    float4 b = ((const float4*)bc)[c4];
    v.x = v.x * d + b.x; v.y = v.y * d + b.y;
    v.z = v.z * d + b.z; v.w = v.w * d + b.w;
    ((float4*)acc)[i4] = v;
}

// One wave per scored edge: dot(out[a], out[b]) over 128 dims.
__global__ void __launch_bounds__(256) k_score(const int* __restrict__ ea,
                                               const int* __restrict__ eb,
                                               const float* __restrict__ out,
                                               float* __restrict__ logits) {
    int e = blockIdx.x * 8 + (threadIdx.x >> 5);
    if (e >= N_PRED) return;
    int lane = threadIdx.x & 31;
    int a = ea[e], b = eb[e];
    float4 va = ((const float4*)out)[a * 32 + lane];
    float4 vb = ((const float4*)out)[b * 32 + lane];
    float p = va.x * vb.x + va.y * vb.y + va.z * vb.z + va.w * vb.w;
#pragma unroll
    for (int off = 16; off > 0; off >>= 1)
        p += __shfl_xor(p, off, 32);
    if (lane == 0) logits[e] = p;
}

// ---------------------------------------------------------------------------
extern "C" void kernel_launch(void* const* d_in, const int* in_sizes, int n_in,
                              void* d_out, int out_size, void* d_ws, size_t ws_size,
                              hipStream_t stream) {
    const float* x   = (const float*)d_in[0];
    const int*   ei  = (const int*)d_in[1];            // [2, N_PRED]
    const int*   pe  = (const int*)d_in[2];            // [2, N_POS]
    const float* W1  = (const float*)d_in[3];
    const float* b1  = (const float*)d_in[4];
    const float* Wc  = (const float*)d_in[5];
    const float* bc  = (const float*)d_in[6];
    float* logits = (float*)d_out;

    const int* pos_row = pe;                           // source
    const int* pos_col = pe + N_POS;                   // target
    const int* sc_a = ei;
    const int* sc_b = ei + N_PRED;

    // workspace layout (~51.7 MB total)
    char* ws = (char*)d_ws;
    bf16_t*   wfrag = (bf16_t*)(ws);                               // 65536 B
    float*    hb    = (float*)(ws + 65536);                        // 512 B
    unsigned* cnt   = (unsigned*)(ws + 66560);                     // 200000 B
    float*    dinv  = (float*)(ws + 66560 + 204800);               // 200000 B
    float*    h     = (float*)(ws + 66560 + 2 * 204800);           // 25.6 MB
    float*    accb  = (float*)(ws + 66560 + 2 * 204800 + 25600000);// 25.6 MB

    // 1) fuse weights into WMMA B fragments
    k_fuse_w<<<128, 256, 0, stream>>>(W1, Wc, wfrag);
    k_fuse_b<<<1, 128, 0, stream>>>(b1, Wc, hb);

    // 2) main GEMM: h = x @ Wf + hb   (3125 M-tiles, 8 waves/block)
    k_gemm<<<391, 256, 0, stream>>>(x, wfrag, hb, h);

    // 3) degree + rsqrt norm
    k_deg_init<<<(N_NODES + 255) / 256, 256, 0, stream>>>(cnt);
    k_deg_acc<<<(N_POS + 255) / 256, 256, 0, stream>>>(pos_col, cnt);
    k_dinv<<<(N_NODES + 255) / 256, 256, 0, stream>>>(cnt, dinv);

    // 4) pre-scale + self-loop init, edge scatter, post-scale + bias
    k_scale<<<(N_NODES * 32) / 256, 256, 0, stream>>>(h, accb, dinv);
    k_scatter<<<(N_POS + 7) / 8, 256, 0, stream>>>(pos_row, pos_col, h, accb);
    k_finish<<<(N_NODES * 32) / 256, 256, 0, stream>>>(accb, dinv, bc);

    // 5) edge dot-product scoring
    k_score<<<(N_PRED + 7) / 8, 256, 0, stream>>>(sc_a, sc_b, accb, logits);
}
